// SSDBoxHead_1460288881514
// MI455X (gfx1250) — compile-verified, matrix operand order checked by reading, not run on hardware
//
#include <hip/hip_runtime.h>
#include <hip/hip_bf16.h>
#include <float.h>

// Problem constants (match reference)
#define B_   32
#define N_   24564
#define C_   81
#define CF_  80      // foreground classes
#define K_   100     // NMS candidate budget
#define KP_  112     // K padded to 7 x 16 WMMA tiles
#define CONF_THR_ 0.01f
#define NMS_THR_  0.45f

typedef __attribute__((ext_vector_type(16))) _Float16 v16h;
typedef __attribute__((ext_vector_type(8)))  float    v8f;

typedef __attribute__((address_space(1))) int* gptr_i32;
typedef __attribute__((address_space(3))) int* lptr_i32;

#if __has_builtin(__builtin_amdgcn_global_load_async_to_lds_b32)
#define HAVE_ASYNC_LDS 1
#else
#define HAVE_ASYNC_LDS 0
#endif

__device__ __forceinline__ void wait_async_then_barrier() {
#if HAVE_ASYNC_LDS
#if __has_builtin(__builtin_amdgcn_s_wait_asynccnt)
  __builtin_amdgcn_s_wait_asynccnt(0);
#else
  asm volatile("s_wait_asynccnt 0" ::: "memory");
#endif
#endif
  __syncthreads();
}

// ---------------------------------------------------------------------------
// Kernel 1: row softmax over C=81. One wave32 per row, 8 rows per block.
// Streams 255 MB in + 255 MB out; lane k covers elements {k, k+32, k+64}.
// ---------------------------------------------------------------------------
__global__ __launch_bounds__(256) void softmax_kernel(
    const float* __restrict__ logits, float* __restrict__ scores) {
  const int lane = threadIdx.x & 31;
  const int wid  = threadIdx.x >> 5;
  const long long row = (long long)blockIdx.x * 8 + wid;
  if (row >= (long long)B_ * N_) return;

  const float* src = logits + row * C_;
  const bool has2 = (lane + 64) < C_;
  float x0 = src[lane];
  float x1 = src[lane + 32];
  float x2 = has2 ? src[lane + 64] : -FLT_MAX;

  float m = fmaxf(fmaxf(x0, x1), x2);
#pragma unroll
  for (int off = 16; off > 0; off >>= 1) m = fmaxf(m, __shfl_xor(m, off, 32));

  float e0 = expf(x0 - m);
  float e1 = expf(x1 - m);
  float e2 = has2 ? expf(x2 - m) : 0.f;
  float s = e0 + e1 + e2;
#pragma unroll
  for (int off = 16; off > 0; off >>= 1) s += __shfl_xor(s, off, 32);
  const float inv = 1.f / s;

  float* dst = scores + row * C_;
  dst[lane]      = e0 * inv;
  dst[lane + 32] = e1 * inv;
  if (has2) dst[lane + 64] = e2 * inv;
}

// ---------------------------------------------------------------------------
// Kernel 2: box decode (center-form regression -> corner form). float4 I/O.
// ---------------------------------------------------------------------------
__global__ __launch_bounds__(256) void decode_kernel(
    const float4* __restrict__ loc, const float4* __restrict__ priors,
    float4* __restrict__ boxes) {
  long long i = (long long)blockIdx.x * blockDim.x + threadIdx.x;
  if (i >= (long long)B_ * N_) return;
  float4 l = loc[i];
  float4 p = priors[i % N_];
  float cx = l.x * 0.1f * p.z + p.x;
  float cy = l.y * 0.1f * p.w + p.y;
  float w  = expf(l.z * 0.2f) * p.z;
  float h  = expf(l.w * 0.2f) * p.w;
  boxes[i] = make_float4(cx - 0.5f * w, cy - 0.5f * h, cx + 0.5f * w, cy + 0.5f * h);
}

// ---------------------------------------------------------------------------
// Kernel 3: per-(image,class) top-100 + greedy NMS. One 256-thread block per
// (b,c). Class column async-gathered into LDS; iterative argmax with
// incremental stripe maxima; pairwise area-sum via WMMA; on-the-fly IoU NMS.
// ---------------------------------------------------------------------------
__global__ __launch_bounds__(256) void topk_nms_kernel(
    const float* __restrict__ scores, const float4* __restrict__ boxes,
    float* __restrict__ nms_scores, float4* __restrict__ nms_boxes,
    float* __restrict__ keep_out) {
  __shared__ union {
    float s[24576];          // staged class column (N_=24564 used)
    float asum[KP_ * KP_];   // reused: pairwise area[i]+area[j] (12544)
  } big;
  __shared__ float  rv[256];
  __shared__ int    ri[256];
  __shared__ float  sel_s[KP_];
  __shared__ int    sel_i[KP_];
  __shared__ float4 cbox[KP_];
  __shared__ float  carea[KP_];
  __shared__ int    kb[KP_];

  const int tid = threadIdx.x;
  const int bc  = blockIdx.x;           // b*80 + c
  const int b   = bc / CF_;
  const int c   = bc % CF_;
  const float* col = scores + (size_t)b * N_ * C_ + (c + 1);

  // ---- Phase 1: stage strided class column (stride 81 floats) into LDS ----
  for (int i = tid; i < N_; i += 256) {
#if HAVE_ASYNC_LDS
    __builtin_amdgcn_global_load_async_to_lds_b32(
        (gptr_i32)(col + (size_t)i * C_),
        (lptr_i32)&big.s[i], 0, 0);
#else
    big.s[i] = col[(size_t)i * C_];
#endif
  }
  wait_async_then_barrier();

  // ---- Phase 2: top-K by iterative argmax with incremental stripe maxima ---
  float bv = -FLT_MAX;
  int   bi = tid;
  for (int i = tid; i < N_; i += 256) {
    float v = big.s[i];
    if (v > bv) { bv = v; bi = i; }
  }
  for (int k = 0; k < K_; ++k) {
    rv[tid] = bv; ri[tid] = bi;
    __syncthreads();
    for (int off = 128; off > 0; off >>= 1) {
      if (tid < off && rv[tid + off] > rv[tid]) {
        rv[tid] = rv[tid + off]; ri[tid] = ri[tid + off];
      }
      __syncthreads();
    }
    const float wv = rv[0];
    const int   wi = ri[0];
    if (tid == 0) { sel_s[k] = wv; sel_i[k] = wi; big.s[wi] = -FLT_MAX; }
    __syncthreads();
    if (tid == (wi & 255)) {            // only the winning stripe rescans
      bv = -FLT_MAX; bi = tid;
      for (int i = tid; i < N_; i += 256) {
        float v = big.s[i];
        if (v > bv) { bv = v; bi = i; }
      }
    }
  }
  __syncthreads();                      // big.s dead after this point

  // ---- Phase 3: gather candidate boxes + areas --------------------------
  if (tid < KP_) {
    float4 bx = make_float4(0.f, 0.f, 0.f, 0.f);
    float  a  = 0.f;
    if (tid < K_) {
      bx = boxes[(size_t)b * N_ + sel_i[tid]];
      a  = fmaxf(bx.z - bx.x, 0.f) * fmaxf(bx.w - bx.y, 0.f);
    }
    cbox[tid]  = bx;
    carea[tid] = a;
  }
  __syncthreads();

  // ---- Phase 4: pairwise area[i]+area[j] via rank-2 WMMA -----------------
  // D = A(16x32) x B(32x16): A row m = [area_i, 1, 0...], B col n = [1, area_j, 0...]
  {
    const int lane = tid & 31;
    const int wid  = tid >> 5;
    const int sub  = lane & 15;
    for (int t = wid; t < 49; t += 8) {           // 7x7 = 49 tiles, uniform per wave
      const int ti = t / 7, tj = t % 7;
      v16h a = {};
      v16h bm = {};
      const float ai = carea[ti * 16 + sub];
      const float aj = carea[tj * 16 + sub];
      if (lane < 16) {                             // lanes 0-15 hold K=0,1 of A; rows K=0,1 of B
        a[0]  = (_Float16)ai;  a[1]  = (_Float16)1.0f;
        bm[0] = (_Float16)1.0f; bm[1] = (_Float16)aj;
      }
      v8f acc = {};
      acc = __builtin_amdgcn_wmma_f32_16x16x32_f16(
          /*neg_a=*/false, a, /*neg_b=*/false, bm,
          /*c_mod=*/(short)0, acc, /*reuse_a=*/false, /*reuse_b=*/false);
      const int mb = (lane < 16) ? 0 : 8;          // D: VGPR r -> M=r (lanes 0-15) / M=8+r
#pragma unroll
      for (int r = 0; r < 8; ++r)
        big.asum[(ti * 16 + mb + r) * KP_ + tj * 16 + sub] = acc[r];
    }
  }
  __syncthreads();

  // ---- Phase 5: greedy NMS (sequential over i, parallel over j) ----------
  if (tid < KP_) kb[tid] = (tid < K_) ? (sel_s[tid] > CONF_THR_ ? 1 : 0) : 0;
  __syncthreads();
  for (int i = 0; i < K_; ++i) {
    if (kb[i]) {                                   // uniform LDS read
      if (tid > i && tid < K_ && kb[tid]) {
        const float4 pi = cbox[i], pj = cbox[tid];
        const float ltx = fmaxf(pi.x, pj.x), lty = fmaxf(pi.y, pj.y);
        const float rbx = fminf(pi.z, pj.z), rby = fminf(pi.w, pj.w);
        const float iw = fmaxf(rbx - ltx, 0.f), ih = fmaxf(rby - lty, 0.f);
        const float inter = iw * ih;
        const float iou = inter / (big.asum[i * KP_ + tid] - inter + 1e-8f);
        if (iou > NMS_THR_) kb[tid] = 0;
      }
    }
    __syncthreads();
  }

  // ---- Phase 6: outputs --------------------------------------------------
  if (tid < K_) {
    const size_t o = (size_t)bc * K_ + tid;
    const int kk = kb[tid];
    nms_scores[o] = kk ? sel_s[tid] : 0.f;
    keep_out[o]   = kk ? 1.f : 0.f;
    nms_boxes[o]  = kk ? cbox[tid] : make_float4(0.f, 0.f, 0.f, 0.f);
  }
}

// ---------------------------------------------------------------------------
extern "C" void kernel_launch(void* const* d_in, const int* in_sizes, int n_in,
                              void* d_out, int out_size, void* d_ws, size_t ws_size,
                              hipStream_t stream) {
  (void)in_sizes; (void)n_in; (void)out_size; (void)d_ws; (void)ws_size;

  const float*  logits = (const float*)d_in[0];
  const float4* loc    = (const float4*)d_in[1];
  const float4* priors = (const float4*)d_in[2];

  float* out = (float*)d_out;
  float*  scores     = out;                                         // B*N*C
  float4* boxes      = (float4*)(out + (size_t)B_ * N_ * C_);       // B*N*4
  float*  nms_scores = out + (size_t)B_ * N_ * C_ + (size_t)B_ * N_ * 4;
  float4* nms_boxes  = (float4*)(nms_scores + (size_t)B_ * CF_ * K_);
  float*  keep       = (float*)nms_boxes + (size_t)B_ * CF_ * K_ * 4;

  const long long rows = (long long)B_ * N_;
  softmax_kernel<<<(int)((rows + 7) / 8), 256, 0, stream>>>(logits, scores);
  decode_kernel <<<(int)((rows + 255) / 256), 256, 0, stream>>>(loc, priors, boxes);
  topk_nms_kernel<<<B_ * CF_, 256, 0, stream>>>(scores, boxes, nms_scores, nms_boxes, keep);
}
